// GNNModel_32890859553003
// MI455X (gfx1250) — matrix operand order repaired and checked by reading, not compiled, not run on hardware
//
#include <hip/hip_runtime.h>
#include <hip/hip_bf16.h>
#include <stdint.h>

typedef __attribute__((ext_vector_type(16))) _Float16     v16h;
typedef __attribute__((ext_vector_type(8)))  float        v8f;
typedef __attribute__((ext_vector_type(4)))  unsigned int v4u;

union Frag16 { v16h h; v4u q[2]; };

// ---------------- degree / normalization ----------------

__global__ void gcn_deg_init(float* deg, int n) {
  int i = blockIdx.x * blockDim.x + threadIdx.x;
  if (i < n) deg[i] = 1.0f;  // self loop contributes 1
}

__global__ void gcn_deg_accum(const int* __restrict__ dst, float* deg, int E) {
  int e = blockIdx.x * blockDim.x + threadIdx.x;
  if (e < E) atomicAdd(&deg[dst[e]], 1.0f);
}

__global__ void gcn_dinv(float* deg, int n) {
  int i = blockIdx.x * blockDim.x + threadIdx.x;
  if (i < n) deg[i] = rsqrtf(deg[i]);  // deg >= 1 always
}

// ---------------- f32 -> f16 conversions (padded K and padded rows) ----------------

// src: [n, K] f32 row-major -> dst: [nPad, Kp] f16, zero padded (rows >= n and k >= K),
// optional ReLU. nPad is a multiple of 16 so the WMMA GEMM needs no row guards.
__global__ void gcn_cvt_h16(const float* __restrict__ src, _Float16* __restrict__ dst,
                            int n, int nPad, int K, int Kp, int relu) {
  size_t t = (size_t)blockIdx.x * blockDim.x + threadIdx.x;
  if (t >= (size_t)nPad * Kp) return;
  int i = (int)(t / Kp);
  int k = (int)(t % Kp);
  float v = (i < n && k < K) ? src[(size_t)i * K + k] : 0.0f;
  if (relu) v = fmaxf(v, 0.0f);
  dst[t] = (_Float16)v;
}

// W: [K, Nout] f32 row-major -> Wt: [Nout, Kp] f16 (transposed, zero padded)
__global__ void gcn_cvt_w16(const float* __restrict__ W, _Float16* __restrict__ Wt,
                            int K, int Nout, int Kp) {
  size_t t = (size_t)blockIdx.x * blockDim.x + threadIdx.x;
  if (t >= (size_t)Nout * Kp) return;
  int c = (int)(t / Kp);
  int k = (int)(t % Kp);
  float v = (k < K) ? W[(size_t)k * Nout + c] : 0.0f;
  Wt[t] = (_Float16)v;
}

// ------ WMMA GEMM: C[MPad,NOUT] = A16[MPad,KP] * Bt16[NOUT,KP]^T, optional row scale ------
// One wave per 16x16 tile; blockDim.x = 32*(NOUT/16); gridDim.x = MPad/16.
// All fragments loaded up front (one clause), then the WMMA chain. Compile-time strides
// so the 8-row store epilogue uses immediate offsets. If SCALE, stores C *= dinv[row]
// (folds the source-side GCN normalization into the GEMM epilogue).

template <int KP, int NOUT, bool SCALE>
__global__ void gcn_wmma_gemm(const _Float16* __restrict__ A,
                              const _Float16* __restrict__ Bt,
                              const float* __restrict__ dinv,
                              float* __restrict__ C) {
  constexpr int NK = KP / 32;
  const int lane = threadIdx.x & 31;
  const int wave = threadIdx.x >> 5;   // tile along NOUT
  const int half = lane >> 4;          // 0: lanes 0-15, 1: lanes 16-31
  const int l    = lane & 15;
  const int tm   = blockIdx.x;
  const int arow = tm * 16 + l;
  const int bcol = wave * 16 + l;

  // A 16x32 f16 fragment: lanes 0-15 hold K {k0..k0+7, k0+16..k0+23} of row arow,
  //                       lanes 16-31 hold K {k0+8..k0+15, k0+24..k0+31}
  // B 32x16 f16 fragment: lanes 0-15 hold K {k0..k0+7, k0+16..k0+23} of column bcol,
  //                       lanes 16-31 hold K {k0+8..k0+15, k0+24..k0+31}
  const _Float16* ap = A  + (size_t)arow * KP + half * 8;
  const _Float16* bp = Bt + (size_t)bcol * KP + half * 8;

  Frag16 a[NK], b[NK];
#pragma unroll
  for (int j = 0; j < NK; ++j) {
    a[j].q[0] = *(const v4u*)(ap + j * 32);
    a[j].q[1] = *(const v4u*)(ap + j * 32 + 16);
    b[j].q[0] = *(const v4u*)(bp + j * 32);
    b[j].q[1] = *(const v4u*)(bp + j * 32 + 16);
  }

  v8f acc = {};
#pragma unroll
  for (int j = 0; j < NK; ++j) {
    acc = __builtin_amdgcn_wmma_f32_16x16x32_f16(
        /*neg_a=*/false, a[j].h, /*neg_b=*/false, b[j].h,
        /*c_mod=*/(short)0, acc, /*reuse_a=*/false, /*reuse_b=*/false);
  }

  // C/D layout: lanes 0-15 -> rows tm*16+0..7, lanes 16-31 -> rows tm*16+8..15; col = l
  const int crow0 = tm * 16 + half * 8;
  float* cp = C + (size_t)crow0 * NOUT + (wave * 16 + l);

  if (SCALE) {
    // dinv[crow0 .. crow0+7], contiguous and 32B aligned
    float4 d0 = *(const float4*)(dinv + crow0);
    float4 d1 = *(const float4*)(dinv + crow0 + 4);
    acc[0] *= d0.x; acc[1] *= d0.y; acc[2] *= d0.z; acc[3] *= d0.w;
    acc[4] *= d1.x; acc[5] *= d1.y; acc[6] *= d1.z; acc[7] *= d1.w;
  }
#pragma unroll
  for (int i = 0; i < 8; ++i)
    cp[(size_t)i * NOUT] = acc[i];   // immediate offsets: i*NOUT*4 bytes
}

// ---------------- propagation (h is pre-scaled by dinv[row] when from GCN GEMM) ----------

// out[i, 4c..4c+3] = hs[i, 4c..4c+3] * dinv[i] + bias[4c..4c+3]
// (self loop: hs = h*dinv[i], so hs*dinv[i] = h*dinv^2; also initializes out)
__global__ void gcn_self_bias(const float4* __restrict__ hs, const float* __restrict__ dinv,
                              const float4* __restrict__ bias, float4* __restrict__ out,
                              int n, int dq) {   // dq = dout/4
  size_t t = (size_t)blockIdx.x * blockDim.x + threadIdx.x;
  if (t >= (size_t)n * dq) return;
  int i = (int)(t / dq);
  int c = (int)(t % dq);
  float s = dinv[i];
  float4 hv = hs[t], bv = bias[c], o;
  o.x = hv.x * s + bv.x;
  o.y = hv.y * s + bv.y;
  o.z = hv.z * s + bv.z;
  o.w = hv.w * s + bv.w;
  out[t] = o;
}

// one thread per (edge, 4 channels): coalesced b128 gather + 4 f32 atomics (L2-resident).
// hs already carries dinv[s]; only dinv[d] needed here.
__global__ void gcn_edge_scatter(const int* __restrict__ src, const int* __restrict__ dst,
                                 const float4* __restrict__ hs, const float* __restrict__ dinv,
                                 float* out, int E, int dq) {   // dq = dout/4
  size_t t = (size_t)blockIdx.x * blockDim.x + threadIdx.x;
  if (t >= (size_t)E * dq) return;
  int e = (int)(t / dq);
  int c = (int)(t % dq);
  int s = src[e];
  int d = dst[e];
  float norm = dinv[d];
  float4 hv = hs[(size_t)s * dq + c];
  float* o = out + ((size_t)d * dq + c) * 4;
  atomicAdd(o + 0, hv.x * norm);
  atomicAdd(o + 1, hv.y * norm);
  atomicAdd(o + 2, hv.z * norm);
  atomicAdd(o + 3, hv.w * norm);
}

// ---------------- MLP head epilogue ----------------
// out[i,:] = relu(hpre[i,:] + fb1) @ fw2 + fb2   (K = 64, 2 outputs)
__global__ void gcn_fc2(const float* __restrict__ hpre, const float* __restrict__ fb1,
                        const float* __restrict__ fw2, const float* __restrict__ fb2,
                        float* __restrict__ out, int n, int K) {
  int i = blockIdx.x * blockDim.x + threadIdx.x;
  if (i >= n) return;
  float a0 = fb2[0], a1 = fb2[1];
  const float* hr = hpre + (size_t)i * K;
  for (int k = 0; k < K; ++k) {
    float v = fmaxf(hr[k] + fb1[k], 0.0f);
    a0 += v * fw2[k * 2 + 0];
    a1 += v * fw2[k * 2 + 1];
  }
  out[(size_t)i * 2 + 0] = a0;
  out[(size_t)i * 2 + 1] = a1;
}

// ---------------- launcher ----------------

extern "C" void kernel_launch(void* const* d_in, const int* in_sizes, int n_in,
                              void* d_out, int out_size, void* d_ws, size_t ws_size,
                              hipStream_t stream) {
  (void)n_in; (void)out_size; (void)ws_size;
  const float* x   = (const float*)d_in[0];
  const int*   ei  = (const int*)d_in[1];
  const float* W1  = (const float*)d_in[2];
  const float* b1  = (const float*)d_in[3];
  const float* W2  = (const float*)d_in[4];
  const float* b2  = (const float*)d_in[5];
  const float* W3  = (const float*)d_in[6];
  const float* b3  = (const float*)d_in[7];
  const float* fw1 = (const float*)d_in[8];
  const float* fb1 = (const float*)d_in[9];
  const float* fw2 = (const float*)d_in[10];
  const float* fb2 = (const float*)d_in[11];

  const int n    = in_sizes[0] / 3;        // 100000
  const int nPad = ((n + 15) / 16) * 16;   // WMMA row padding
  const int E    = in_sizes[1] / 2;        // 600000
  const int* src = ei;                     // edge_index row 0
  const int* dst = ei + E;                 // edge_index row 1

  auto al = [](size_t v) { return (v + 255) & ~(size_t)255; };
  char* ws = (char*)d_ws;
  size_t off = 0;
  float*    dinv = (float*)(ws + off);    off += al((size_t)(nPad + 16) * 4);
  float*    bufA = (float*)(ws + off);    off += al((size_t)nPad * 128 * 4);
  float*    bufB = (float*)(ws + off);    off += al((size_t)nPad * 128 * 4);
  _Float16* h16  = (_Float16*)(ws + off); off += al((size_t)nPad * 128 * 2);
  _Float16* w16  = (_Float16*)(ws + off); off += al((size_t)128 * 128 * 2);

  const int T = 256;

  // --- degree normalization (shared by all 3 GCN layers); pad rows get deg=1 -> dinv=1 ---
  gcn_deg_init <<<(nPad + T - 1) / T, T, 0, stream>>>(dinv, nPad);
  gcn_deg_accum<<<(E + T - 1) / T, T, 0, stream>>>(dst, dinv, E);
  gcn_dinv     <<<(nPad + T - 1) / T, T, 0, stream>>>(dinv, nPad);

  auto cvt_h = [&](const float* s, int K, int Kp, int relu) {
    size_t tot = (size_t)nPad * Kp;
    gcn_cvt_h16<<<(unsigned)((tot + T - 1) / T), T, 0, stream>>>(s, h16, n, nPad, K, Kp, relu);
  };
  auto cvt_w = [&](const float* W, int K, int Nout, int Kp) {
    size_t tot = (size_t)Nout * Kp;
    gcn_cvt_w16<<<(unsigned)((tot + T - 1) / T), T, 0, stream>>>(W, w16, K, Nout, Kp);
  };
  auto prop = [&](const float* hpre, const float* bias, float* outbuf, int dout) {
    int dq = dout / 4;
    size_t tot = (size_t)n * dq;
    gcn_self_bias<<<(unsigned)((tot + T - 1) / T), T, 0, stream>>>(
        (const float4*)hpre, dinv, (const float4*)bias, (float4*)outbuf, n, dq);
    size_t tote = (size_t)E * dq;
    gcn_edge_scatter<<<(unsigned)((tote + T - 1) / T), T, 0, stream>>>(
        src, dst, (const float4*)hpre, dinv, outbuf, E, dq);
  };

  dim3 g(nPad / 16);

  // --- GCN layer 1: 3 -> 64 (K padded to 32); GEMM stores h * dinv[row] ---
  cvt_h(x, 3, 32, 0);
  cvt_w(W1, 3, 64, 32);
  gcn_wmma_gemm<32, 64, true><<<g, dim3(32 * 4), 0, stream>>>(h16, w16, dinv, bufA);
  prop(bufA, b1, bufB, 64);
  cvt_h(bufB, 64, 64, 1);   // ReLU + f16 for next GEMM

  // --- GCN layer 2: 64 -> 128 ---
  cvt_w(W2, 64, 128, 64);
  gcn_wmma_gemm<64, 128, true><<<g, dim3(32 * 8), 0, stream>>>(h16, w16, dinv, bufA);
  prop(bufA, b2, bufB, 128);
  cvt_h(bufB, 128, 128, 1);

  // --- GCN layer 3: 128 -> 128 ---
  cvt_w(W3, 128, 128, 128);
  gcn_wmma_gemm<128, 128, true><<<g, dim3(32 * 8), 0, stream>>>(h16, w16, dinv, bufA);
  prop(bufA, b3, bufB, 128);
  cvt_h(bufB, 128, 128, 1);

  // --- fc1: 128 -> 64, no norm scale (bias+ReLU fused into fc2 epilogue) ---
  cvt_w(fw1, 128, 64, 128);
  gcn_wmma_gemm<128, 64, false><<<g, dim3(32 * 4), 0, stream>>>(h16, w16, dinv, bufA);

  // --- fc2: 64 -> 2 ---
  gcn_fc2<<<(n + T - 1) / T, T, 0, stream>>>(bufA, fb1, fw2, fb2, (float*)d_out, n, 64);
}
